// MultiHeadedAttention_82042465288446
// MI455X (gfx1250) — matrix operand order repaired and checked by reading, not compiled, not run on hardware
//
#include <hip/hip_runtime.h>
#include <hip/hip_bf16.h>

#define H 8
#define DM 512
#define SEQ 1024
#define BATCH 16
#define KS 7
#define DK 64
#define KDIM (SEQ * KS) /* 7168 */

typedef __attribute__((ext_vector_type(16))) __bf16 v16bf;
typedef __attribute__((ext_vector_type(8)))  float  v8f;
typedef unsigned int v4u __attribute__((ext_vector_type(4)));
typedef int v8i __attribute__((ext_vector_type(8)));
typedef int v4i __attribute__((ext_vector_type(4)));
typedef __bf16 bf16;

union BF16x8 { uint4 u; bf16 h[8]; };

__device__ __forceinline__ v8f wmma_bf16(v16bf a, v16bf b, v8f c) {
  return __builtin_amdgcn_wmma_f32_16x16x32_bf16(false, a, false, b, (short)0, c,
                                                 false, false);
}

// 8 bf16 from p0 -> elems 0..7, 8 bf16 from p1 -> elems 8..15 (both 16B aligned)
__device__ __forceinline__ v16bf load_frag16(const bf16* p0, const bf16* p1) {
  union { v16bf v; uint4 u[2]; } t;
  t.u[0] = *(const uint4*)p0;
  t.u[1] = *(const uint4*)p1;
  return t.v;
}

// ---------------- Tensor Data Mover: 2-D bf16 tile -> (optionally padded) LDS -----
// D# layout per CDNA5 ISA. data_size code 1 (2 bytes).
// pad codes: interval c -> 2^(c+1) dwords between pads; amount c -> (c+1) dwords.
__device__ __forceinline__ void tdm_load_tile_2d(unsigned lds_off, const void* gaddr,
                                                 unsigned tile_d0, unsigned tile_d1,
                                                 unsigned tensor_d0, unsigned tensor_d1,
                                                 unsigned long stride0_elems,
                                                 unsigned pad_en, unsigned pad_int_c,
                                                 unsigned pad_amt_c) {
  unsigned long ga = (unsigned long)gaddr;
  v4u g0;
  g0.x = 1u;                                              // count=1, user mode
  g0.y = lds_off;                                         // bits 63:32
  g0.z = (unsigned)ga;                                    // global_addr[31:0]
  g0.w = (unsigned)((ga >> 32) & 0x1FFFFFFu) | (2u << 30);// addr[56:32] | type=2
  v8i g1;
  g1[0] = (int)((1u << 16) | (pad_en << 20) | (pad_int_c << 22) | (pad_amt_c << 25));
  g1[1] = (int)((tensor_d0 & 0xFFFFu) << 16);             // tensor_dim0[15:0] @ bit48
  g1[2] = (int)((tensor_d0 >> 16) | ((tensor_d1 & 0xFFFFu) << 16));
  g1[3] = (int)((tensor_d1 >> 16) | (tile_d0 << 16));     // tile_dim0 @ bit112
  g1[4] = (int)(tile_d1);                                 // tile_dim1, tile_dim2=0
  g1[5] = (int)(unsigned)(stride0_elems & 0xFFFFFFFFu);
  g1[6] = (int)(unsigned)(stride0_elems >> 32);           // dim1_stride=0 (2-D)
  g1[7] = 0;
  v4i gz = {0, 0, 0, 0};
#if defined(__clang_major__) && (__clang_major__ >= 23)
  v8i gz8 = {0, 0, 0, 0, 0, 0, 0, 0};
  __builtin_amdgcn_tensor_load_to_lds(g0, g1, gz, gz, gz8, 0);
#else
  __builtin_amdgcn_tensor_load_to_lds(g0, g1, gz, gz, 0);
#endif
}

// ---------------- weight repack: f32 [O,I,T] -> bf16 [O, t*1024+i] ----------------
__global__ void pack_w_kernel(const float* __restrict__ w, bf16* __restrict__ wp) {
  long idx = (long)blockIdx.x * blockDim.x + threadIdx.x; // < 1024*7168
  int o = (int)(idx / KDIM);
  int rem = (int)(idx - (long)o * KDIM);
  int t = rem >> 10;
  int i = rem & 1023;
  wp[idx] = (bf16)w[(long)o * KDIM + i * KS + t];
}

// ---------------- elementwise f32 -> bf16 ----------------
__global__ void cvt_bf16_kernel(const float* __restrict__ x, bf16* __restrict__ y, long n) {
  long i = (long)blockIdx.x * blockDim.x + threadIdx.x;
  if (i < n) y[i] = (bf16)x[i];
}

// ---------------- conv-as-GEMM: Y[b,o,l] = sum_k Wp[o,k] * x[b,i,l+t-6] ----------------
// Double-buffered: TDM pulls weight tile i+1 while WMMAs consume tile i.
#define CONV_MT 64
#define CONV_NT 128
#define LPAD 40 /* 32 + 8 pad, keeps 16B alignment (80B row) */

__global__ __launch_bounds__(256)
void conv_gemm_kernel(const float* __restrict__ x,    // [B,1024,512] f32
                      const bf16*  __restrict__ wp,   // [1024,7168] bf16 packed
                      const float* __restrict__ bias, // [1024]
                      bf16* __restrict__ y)           // [B,1024,512] bf16
{
  __shared__ bf16 Al[2][CONV_MT][LPAD];   // A tiles [m][k]  (TDM-filled, padded rows)
  __shared__ bf16 Bl[2][CONV_NT][LPAD];   // B tiles transposed [n][k]

  int bt = blockIdx.x;
  int b  = bt >> 6;
  int r  = bt & 63;
  int m0 = (r >> 2) * CONV_MT;  // 16 m-tiles
  int n0 = (r & 3)  * CONV_NT;  // 4 n-tiles

  int tid = threadIdx.x, lane = tid & 31, wave = tid >> 5;
  int half = lane >> 4, l15 = lane & 15;
  int m_off = (wave & 3) * 16;
  int n_off = (wave >> 2) * 64;

  unsigned al_off[2] = { (unsigned)(size_t)&Al[0][0][0],
                         (unsigned)(size_t)&Al[1][0][0] };

  v8f acc[4];
  #pragma unroll
  for (int i = 0; i < 4; ++i) acc[i] = {};

  const long xb = (long)b * SEQ * DM;
  int bn = tid >> 1;            // staged output column (0..127)
  int bk = (tid & 1) * 16;      // k sub-chunk (0 or 16)

  auto issue_a = [&](int k0, int buf) {
    tdm_load_tile_2d(al_off[buf], wp + (long)m0 * KDIM + k0,
                     /*tile*/32, CONV_MT, /*tensor*/KDIM, SEQ,
                     (unsigned long)KDIM, /*pad*/1, 3, 3);
  };
  auto stage_b = [&](int k0, int buf) {
    int tsh = k0 >> 10;     // conv tap t (uniform over the 32-wide k slice)
    int i0  = k0 & 1023;    // input-channel base
    int lsrc = n0 + bn + tsh - 6;   // invariant per thread -> one branch
    float v[16];
    if (lsrc >= 0) {
      const float* src = x + xb + (long)(i0 + bk) * DM + lsrc;
      #pragma unroll
      for (int j = 0; j < 16; ++j) v[j] = src[(long)j * DM];
    } else {
      #pragma unroll
      for (int j = 0; j < 16; ++j) v[j] = 0.0f;
    }
    BF16x8 lo, hi;
    #pragma unroll
    for (int j = 0; j < 8; ++j) { lo.h[j] = (bf16)v[j]; hi.h[j] = (bf16)v[8 + j]; }
    *(uint4*)&Bl[buf][bn][bk]     = lo.u;
    *(uint4*)&Bl[buf][bn][bk + 8] = hi.u;
  };

  // prologue: fill buffer 0
  if (wave == 0) issue_a(0, 0);
  stage_b(0, 0);
  if (wave == 0) __builtin_amdgcn_s_wait_tensorcnt(0);
  __syncthreads();

  int cur = 0;
  for (int k0 = 0; k0 < KDIM; k0 += 32) {
    int nxt = cur ^ 1;
    if (k0 + 32 < KDIM) {            // stage next tile while computing current
      if (wave == 0) issue_a(k0 + 32, nxt);
      stage_b(k0 + 32, nxt);
    }

    v16bf af = load_frag16(&Al[cur][m_off + l15][half * 16],
                           &Al[cur][m_off + l15][8 + half * 16]);
    const bf16* b0 = &Bl[cur][n_off + l15][16 * half];
    const bf16* b1 = &Bl[cur][n_off + 16 + l15][16 * half];
    const bf16* b2 = &Bl[cur][n_off + 32 + l15][16 * half];
    const bf16* b3 = &Bl[cur][n_off + 48 + l15][16 * half];
    v16bf bf0 = load_frag16(b0, b0 + 8);
    v16bf bf1 = load_frag16(b1, b1 + 8);
    v16bf bf2 = load_frag16(b2, b2 + 8);
    v16bf bf3 = load_frag16(b3, b3 + 8);
    acc[0] = wmma_bf16(af, bf0, acc[0]);
    acc[1] = wmma_bf16(af, bf1, acc[1]);
    acc[2] = wmma_bf16(af, bf2, acc[2]);
    acc[3] = wmma_bf16(af, bf3, acc[3]);

    if (wave == 0) __builtin_amdgcn_s_wait_tensorcnt(0);
    __syncthreads();
    cur = nxt;
  }

  #pragma unroll
  for (int nc = 0; nc < 4; ++nc) {
    #pragma unroll
    for (int rr = 0; rr < 8; ++rr) {
      int o = m0 + m_off + rr + 8 * half;   // D: M = r + 8*(lane>=16)
      int l = n0 + n_off + nc * 16 + l15;   // N = lane&15
      y[((long)b * SEQ + o) * DM + l] = (bf16)(acc[nc][rr] + bias[o]);
    }
  }
}

// ---------------- flash attention, one (b,h,128-row q block) per WG ----------------
// Double-buffered K (TDM) / V (cooperative transpose) blocks.
#define KBLK 32

__global__ __launch_bounds__(256)
void attn_kernel(const bf16* __restrict__ Q,  // [B,1024,512] bf16 (head h at col h*64)
                 const bf16* __restrict__ K,
                 const bf16* __restrict__ V,
                 bf16* __restrict__ O)        // [B,1024,512] bf16
{
  __shared__ bf16 Kl[2][KBLK][72];      // [krow][d], TDM-filled with 4-dword pads
  __shared__ bf16 Vl[2][DK][LPAD];      // transposed [d][krow]
  __shared__ bf16 Pl[8][16][LPAD];      // per-wave P tile [row][n]

  int xid = blockIdx.x;
  int b  = xid >> 6;
  int h  = (xid >> 3) & 7;
  int qb = xid & 7;

  int tid = threadIdx.x, lane = tid & 31, wave = tid >> 5;
  int half = lane >> 4, l15 = lane & 15;

  const long base = (long)b * SEQ * DM + h * DK;
  int qrow = qb * 128 + wave * 16 + l15;    // A-fragment row M = lane&15

  unsigned kl_off[2] = { (unsigned)(size_t)&Kl[0][0][0],
                         (unsigned)(size_t)&Kl[1][0][0] };

  const bf16* qp = Q + base + (long)qrow * DM;
  v16bf qa0 = load_frag16(qp + half * 16,      qp + 8 + half * 16);  // k 0..31
  v16bf qa1 = load_frag16(qp + 32 + half * 16, qp + 40 + half * 16);  // k 32..63

  v8f o0 = {}, o1 = {}, o2 = {}, o3 = {};
  v8f m_run, l_run;
  #pragma unroll
  for (int rr = 0; rr < 8; ++rr) { m_run[rr] = -3.0e38f; l_run[rr] = 0.0f; }

  auto issue_k = [&](int kb, int buf) {
    tdm_load_tile_2d(kl_off[buf], K + base + (long)(kb * KBLK) * DM,
                     /*tile*/DK, KBLK, /*tensor*/DM, SEQ,
                     (unsigned long)DM, /*pad*/1, 4, 3);
  };
  auto stage_v = [&](int kb, int buf) {
    int kk = tid >> 3;
    int ck = (tid & 7) * 8;
    const bf16* vs = V + base + (long)(kb * KBLK + kk) * DM + ck;
    BF16x8 vv; vv.u = *(const uint4*)vs;
    #pragma unroll
    for (int j = 0; j < 8; ++j) Vl[buf][ck + j][kk] = vv.h[j];
  };

  // prologue: fill buffer 0
  if (wave == 0) issue_k(0, 0);
  stage_v(0, 0);
  if (wave == 0) __builtin_amdgcn_s_wait_tensorcnt(0);
  __syncthreads();

  int cur = 0;
  for (int kb = 0; kb < SEQ / KBLK; ++kb) {
    int nxt = cur ^ 1;
    if (kb + 1 < SEQ / KBLK) {       // stage next K/V block during compute
      if (wave == 0) issue_k(kb + 1, nxt);
      stage_v(kb + 1, nxt);
    }

    // S = Q K^T for 32 keys -> two 16x16 f32 accumulators
    const bf16* p00 = &Kl[cur][l15][16 * half];
    const bf16* p10 = &Kl[cur][l15][32 + 16 * half];
    const bf16* p01 = &Kl[cur][16 + l15][16 * half];
    const bf16* p11 = &Kl[cur][16 + l15][32 + 16 * half];
    v16bf b00 = load_frag16(p00, p00 + 8);
    v16bf b10 = load_frag16(p10, p10 + 8);
    v16bf b01 = load_frag16(p01, p01 + 8);
    v16bf b11 = load_frag16(p11, p11 + 8);
    v8f s0 = {}, s1 = {};
    s0 = wmma_bf16(qa0, b00, s0);
    s0 = wmma_bf16(qa1, b10, s0);
    s1 = wmma_bf16(qa0, b01, s1);
    s1 = wmma_bf16(qa1, b11, s1);

    const float scale = 0.125f; // 1/sqrt(64)
    v8f corr;
    #pragma unroll
    for (int rr = 0; rr < 8; ++rr) {
      s0[rr] *= scale; s1[rr] *= scale;
      float bm = fmaxf(s0[rr], s1[rr]);        // row lives in one 16-lane half
      bm = fmaxf(bm, __shfl_xor(bm, 1, 32));
      bm = fmaxf(bm, __shfl_xor(bm, 2, 32));
      bm = fmaxf(bm, __shfl_xor(bm, 4, 32));
      bm = fmaxf(bm, __shfl_xor(bm, 8, 32));
      float nm = fmaxf(m_run[rr], bm);
      corr[rr] = __expf(m_run[rr] - nm);
      m_run[rr] = nm;
      s0[rr] = __expf(s0[rr] - nm);
      s1[rr] = __expf(s1[rr] - nm);
      float rs = s0[rr] + s1[rr];
      rs += __shfl_xor(rs, 1, 32);
      rs += __shfl_xor(rs, 2, 32);
      rs += __shfl_xor(rs, 4, 32);
      rs += __shfl_xor(rs, 8, 32);
      l_run[rr] = l_run[rr] * corr[rr] + rs;
    }

    // round-trip P through wave-private LDS to re-lay out as an A fragment
    #pragma unroll
    for (int rr = 0; rr < 8; ++rr) {
      int row = rr + 8 * half;
      Pl[wave][row][l15]      = (bf16)s0[rr];
      Pl[wave][row][16 + l15] = (bf16)s1[rr];
    }
    v16bf pa = load_frag16(&Pl[wave][l15][half * 16],
                           &Pl[wave][l15][8 + half * 16]);

    #pragma unroll
    for (int rr = 0; rr < 8; ++rr) {
      o0[rr] *= corr[rr]; o1[rr] *= corr[rr];
      o2[rr] *= corr[rr]; o3[rr] *= corr[rr];
    }
    const bf16* v0p = &Vl[cur][l15][16 * half];
    const bf16* v1p = &Vl[cur][16 + l15][16 * half];
    const bf16* v2p = &Vl[cur][32 + l15][16 * half];
    const bf16* v3p = &Vl[cur][48 + l15][16 * half];
    v16bf vf0 = load_frag16(v0p, v0p + 8);
    v16bf vf1 = load_frag16(v1p, v1p + 8);
    v16bf vf2 = load_frag16(v2p, v2p + 8);
    v16bf vf3 = load_frag16(v3p, v3p + 8);
    o0 = wmma_bf16(pa, vf0, o0);
    o1 = wmma_bf16(pa, vf1, o1);
    o2 = wmma_bf16(pa, vf2, o2);
    o3 = wmma_bf16(pa, vf3, o3);

    if (wave == 0) __builtin_amdgcn_s_wait_tensorcnt(0);
    __syncthreads();
    cur = nxt;
  }

  int qout = qb * 128 + wave * 16;
  #pragma unroll
  for (int rr = 0; rr < 8; ++rr) {
    float inv = 1.0f / l_run[rr];
    bf16* op = O + base + (long)(qout + rr + 8 * half) * DM;
    op[l15]      = (bf16)(o0[rr] * inv);
    op[16 + l15] = (bf16)(o1[rr] * inv);
    op[32 + l15] = (bf16)(o2[rr] * inv);
    op[48 + l15] = (bf16)(o3[rr] * inv);
  }
}

// ---------------- output linear: out = X @ W^T + b (direct-from-global WMMA) -------
__global__ __launch_bounds__(256)
void linear_kernel(const bf16* __restrict__ Xb,   // [16384, 512]
                   const bf16* __restrict__ Wb,   // [512(out), 512(in)]
                   const float* __restrict__ bias,
                   float* __restrict__ out)       // [16384, 512] f32
{
  int xid = blockIdx.x;
  int rt = xid >> 3;  // 128-row tiles
  int nt = xid & 7;   // 64-col tiles
  int tid = threadIdx.x, lane = tid & 31, wave = tid >> 5;
  int half = lane >> 4, l15 = lane & 15;

  int row = rt * 128 + wave * 16 + l15;
  int n0  = nt * 64;
  const bf16* xp = Xb + (long)row * DM;

  v8f acc[4];
  #pragma unroll
  for (int i = 0; i < 4; ++i) acc[i] = {};

  #pragma unroll 4
  for (int k0 = 0; k0 < DM; k0 += 32) {
    v16bf af = load_frag16(xp + k0 + half * 16, xp + k0 + 8 + half * 16);
    const bf16* w0 = Wb + (long)(n0 + l15) * DM + k0 + 16 * half;
    const bf16* w1 = Wb + (long)(n0 + 16 + l15) * DM + k0 + 16 * half;
    const bf16* w2 = Wb + (long)(n0 + 32 + l15) * DM + k0 + 16 * half;
    const bf16* w3 = Wb + (long)(n0 + 48 + l15) * DM + k0 + 16 * half;
    v16bf bf0 = load_frag16(w0, w0 + 8);
    v16bf bf1 = load_frag16(w1, w1 + 8);
    v16bf bf2 = load_frag16(w2, w2 + 8);
    v16bf bf3 = load_frag16(w3, w3 + 8);
    acc[0] = wmma_bf16(af, bf0, acc[0]);
    acc[1] = wmma_bf16(af, bf1, acc[1]);
    acc[2] = wmma_bf16(af, bf2, acc[2]);
    acc[3] = wmma_bf16(af, bf3, acc[3]);
  }

  int rbase = rt * 128 + wave * 16;
  #pragma unroll
  for (int nc = 0; nc < 4; ++nc) {
    int o = n0 + nc * 16 + l15;
    float bv = bias[o];
    #pragma unroll
    for (int rr = 0; rr < 8; ++rr) {
      out[(long)(rbase + rr + 8 * half) * DM + o] = acc[nc][rr] + bv;
    }
  }
}

// ---------------- launch ----------------
extern "C" void kernel_launch(void* const* d_in, const int* in_sizes, int n_in,
                              void* d_out, int out_size, void* d_ws, size_t ws_size,
                              hipStream_t stream) {
  const float* query    = (const float*)d_in[0];
  const float* key      = (const float*)d_in[1];
  const float* value    = (const float*)d_in[2];
  const float* conv_k_w = (const float*)d_in[3];
  const float* conv_k_b = (const float*)d_in[4];
  const float* conv_v_w = (const float*)d_in[5];
  const float* conv_v_b = (const float*)d_in[6];
  const float* lin_w    = (const float*)d_in[7];
  const float* lin_b    = (const float*)d_in[8];
  (void)in_sizes; (void)n_in; (void)out_size; (void)ws_size;

  char* ws = (char*)d_ws;
  const size_t W_BYTES = (size_t)SEQ * KDIM * sizeof(bf16);      // 14,680,064
  const size_t T_BYTES = (size_t)BATCH * SEQ * DM * sizeof(bf16);// 16,777,216
  bf16* wkp = (bf16*)(ws); ws += W_BYTES;
  bf16* wvp = (bf16*)(ws); ws += W_BYTES;
  bf16* qb  = (bf16*)(ws); ws += T_BYTES;
  bf16* kb  = (bf16*)(ws); ws += T_BYTES;
  bf16* vb  = (bf16*)(ws); ws += T_BYTES;
  bf16* xb  = (bf16*)(ws); ws += T_BYTES;
  bf16* wlb = (bf16*)(ws); ws += (size_t)DM * DM * sizeof(bf16);

  // 1) repack conv weights to bf16 [o, t*1024+i]
  pack_w_kernel<<<(SEQ * KDIM) / 256, 256, 0, stream>>>(conv_k_w, wkp);
  pack_w_kernel<<<(SEQ * KDIM) / 256, 256, 0, stream>>>(conv_v_w, wvp);
  // 2) bf16 conversions
  cvt_bf16_kernel<<<(BATCH * SEQ * DM) / 256, 256, 0, stream>>>(query, qb,
                                                                (long)BATCH * SEQ * DM);
  cvt_bf16_kernel<<<(DM * DM) / 256, 256, 0, stream>>>(lin_w, wlb, (long)DM * DM);
  // 3) conv projections as WMMA GEMMs (double-buffered, TDM-fed A tiles)
  conv_gemm_kernel<<<BATCH * 16 * 4, 256, 0, stream>>>(key, wkp, conv_k_b, kb);
  conv_gemm_kernel<<<BATCH * 16 * 4, 256, 0, stream>>>(value, wvp, conv_v_b, vb);
  // 4) flash attention (double-buffered, TDM-fed K tiles)
  attn_kernel<<<BATCH * H * (SEQ / 128), 256, 0, stream>>>(qb, kb, vb, xb);
  // 5) output linear
  linear_kernel<<<(BATCH * SEQ / 128) * (DM / 64), 256, 0, stream>>>(xb, wlb, lin_b,
                                                                     (float*)d_out);
}
